// TAMBlock_58574763983349
// MI455X (gfx1250) — matrix operand order repaired and checked by reading, not compiled, not run on hardware
//
#include <hip/hip_runtime.h>
#include <math.h>
#include <stdint.h>

typedef __attribute__((ext_vector_type(16))) _Float16 v16h;
typedef __attribute__((ext_vector_type(8)))  float    v8f;
typedef __attribute__((ext_vector_type(4)))  float    v4f;

#define BB 8
#define CC 64
#define HH 256
#define WW 256
#define HW (HH * WW)

#define USE_ASYNC_LDS 1

// ---------------------------------------------------------------------------
// Reduction over C: m3[b][0][h][w] = max_c x, m3[b][1][h][w] = mean_c x
// ---------------------------------------------------------------------------
__global__ void reduce_c_kernel(const float* __restrict__ x, float* __restrict__ m3) {
    int idx = blockIdx.x * blockDim.x + threadIdx.x;      // b*HW + hw
    if (idx >= BB * HW) return;
    int b  = idx >> 16;
    int hw = idx & (HW - 1);
    const float* p = x + (size_t)b * CC * HW + hw;
    float mx = -__builtin_inff();
    float sm = 0.0f;
#pragma unroll 8
    for (int c = 0; c < CC; ++c) {
        float v = p[(size_t)c * HW];
        mx = fmaxf(mx, v);
        sm += v;
    }
    m3[(size_t)b * 2 * HW + hw]      = mx;
    m3[(size_t)b * 2 * HW + HW + hw] = sm * (1.0f / (float)CC);
}

// ---------------------------------------------------------------------------
// Reduction over H: m1[b][0][c][w] = max_h x, m1[b][1][c][w] = mean_h x
// ---------------------------------------------------------------------------
__global__ void reduce_h_kernel(const float* __restrict__ x, float* __restrict__ m1) {
    int idx = blockIdx.x * blockDim.x + threadIdx.x;      // b*(C*W) + c*W + w
    if (idx >= BB * CC * WW) return;
    int b = idx >> 14;                 // C*W = 16384
    int r = idx & 16383;
    int c = r >> 8;
    int w = r & 255;
    const float* p = x + ((size_t)(b * CC + c) * HH) * WW + w;
    float mx = -__builtin_inff();
    float sm = 0.0f;
#pragma unroll 8
    for (int h = 0; h < HH; ++h) {
        float v = p[(size_t)h * WW];
        mx = fmaxf(mx, v);
        sm += v;
    }
    m1[((size_t)(b * 2 + 0) * CC + c) * WW + w] = mx;
    m1[((size_t)(b * 2 + 1) * CC + c) * WW + w] = sm * (1.0f / (float)HH);
}

// ---------------------------------------------------------------------------
// Reduction over W: m2[b][0][h][c] = max_w x, m2[b][1][h][c] = mean_w x
// one wave32 per (b,c,h) row; wave32 __shfl_xor tree.
// ---------------------------------------------------------------------------
__global__ void reduce_w_kernel(const float* __restrict__ x, float* __restrict__ m2) {
    int gwave = (blockIdx.x * blockDim.x + threadIdx.x) >> 5;   // 0 .. B*C*H-1
    int lane  = threadIdx.x & 31;
    if (gwave >= BB * CC * HH) return;
    int b = gwave >> 14;               // C*H = 16384
    int r = gwave & 16383;
    int c = r >> 8;
    int h = r & 255;
    const float* p = x + (size_t)((b * CC + c) * HH + h) * WW;
    float mx = -__builtin_inff();
    float sm = 0.0f;
#pragma unroll
    for (int i = lane; i < WW; i += 32) {
        float v = p[i];
        mx = fmaxf(mx, v);
        sm += v;
    }
#pragma unroll
    for (int off = 16; off >= 1; off >>= 1) {
        mx = fmaxf(mx, __shfl_xor(mx, off, 32));
        sm += __shfl_xor(sm, off, 32);
    }
    if (lane == 0) {
        m2[((size_t)(b * 2 + 0) * HH + h) * CC + c] = mx;
        m2[((size_t)(b * 2 + 1) * HH + h) * CC + c] = sm * (1.0f / (float)WW);
    }
}

// ---------------------------------------------------------------------------
// 7x7 conv (2 -> 1 channel, SAME) + bias + sigmoid, via WMMA implicit GEMM.
// K = 2*7*7 = 98, padded to 128 -> 4 chained v_wmma_f32_16x16x32_f16.
// Templated on plane shape so all block decode is shifts (no SALU division).
// Tile staging uses global_load_async_to_lds_b32 (gfx1250 async path).
// ---------------------------------------------------------------------------
template <int Ph, int Pw, int PPB>
__global__ void conv_wmma_kernel(const float* __restrict__ m,
                                 const float* __restrict__ wgt,
                                 const float* __restrict__ bias,
                                 float* __restrict__ s) {
    constexpr int TW  = PPB + 6;
    constexpr int TOT = 2 * 7 * TW;
    __shared__ float    tile[TOT];
    __shared__ _Float16 wh[128];
    __shared__ int      koff[128];

    const int tid = threadIdx.x;
    constexpr int NTHREADS = PPB * 2;       // PPB/16 waves x 32 lanes

    // decode block -> (batch, output row y, strip start x0); all powers of two
    constexpr int BPR = Pw / PPB;           // blocks per row
    int blk = blockIdx.x;
    int bb  = blk / (Ph * BPR);
    int rr  = blk % (Ph * BPR);
    int y   = rr / BPR;
    int x0  = (rr % BPR) * PPB;

    // stage weights (as f16) and K -> tile-offset LUT
    for (int k = tid; k < 128; k += NTHREADS) {
        if (k < 98) {
            wh[k] = (_Float16)wgt[k];
            int ic = k / 49, r2 = k % 49, ky = r2 / 7, kx = r2 % 7;
            koff[k] = (ic * 7 + ky) * TW + kx;
        } else {
            wh[k] = (_Float16)0.0f;
            koff[k] = -1;
        }
    }

    // stage 2 x 7 x TW input tile with zero padding at plane borders
    const float* mb = m + (size_t)bb * 2 * Ph * Pw;
    for (int t = tid; t < TOT; t += NTHREADS) {
        int ic = t / (7 * TW);
        int r2 = t % (7 * TW);
        int ty = r2 / TW;
        int tx = r2 % TW;
        int gy = y + ty - 3;
        int gx = x0 + tx - 3;
        if (gy >= 0 && gy < Ph && gx >= 0 && gx < Pw) {
#if USE_ASYNC_LDS
            // async copy global -> LDS (ASYNCcnt path); generic LDS pointer's
            // low 32 bits are the LDS byte offset (ISA 10.2 aperture mapping)
            uint64_t ga = (uint64_t)(uintptr_t)&mb[((size_t)ic * Ph + gy) * Pw + gx];
            uint32_t la = (uint32_t)(uintptr_t)&tile[t];
            asm volatile("global_load_async_to_lds_b32 %0, %1, off"
                         :: "v"(la), "v"(ga) : "memory");
#else
            tile[t] = mb[((size_t)ic * Ph + gy) * Pw + gx];
#endif
        } else {
            tile[t] = 0.0f;
        }
    }
#if USE_ASYNC_LDS
    asm volatile("s_wait_asynccnt 0x0" ::: "memory");
#endif
    __syncthreads();

    const int wave = tid >> 5;
    const int lane = tid & 31;
    const int p    = wave * 16 + (lane & 15);   // A-matrix row = pixel in strip
    const int hi   = lane >> 4;                 // lane group (0: 0-15, 1: 16-31)

    v8f acc = {};
#pragma unroll
    for (int kk = 0; kk < 4; ++kk) {
        v16h a, bf;
#pragma unroll
        for (int j = 0; j < 16; ++j) {
            // 16-bit A 16x32 layout: K = j (+8 if j>=8) (+8 for lanes 16-31)
            int ka = kk * 32 + j + ((j >= 8) ? 8 : 0) + (hi ? 8 : 0);
            // 16-bit B 32x16 layout: same K order as A: K = j (+8 if j>=8) (+8 for lanes 16-31)
            int kb = kk * 32 + j + ((j >= 8) ? 8 : 0) + (hi ? 8 : 0);
            int off = koff[ka];
            a[j]  = (off >= 0) ? (_Float16)tile[off + p] : (_Float16)0.0f;
            bf[j] = wh[kb];
        }
        acc = __builtin_amdgcn_wmma_f32_16x16x32_f16(
            /*neg_a=*/false, a, /*neg_b=*/false, bf,
            /*c_mod=*/(short)0, acc, /*reuse_a=*/false, /*reuse_b=*/false);
    }

    // C/D layout: VGPR r holds row M=r (lanes 0-15) / M=8+r (lanes 16-31),
    // col N = lane&15. All 16 columns identical -> column-0 lanes write 8 rows.
    if ((lane & 15) == 0) {
        float bv = bias[0];
        float* sb = s + (size_t)bb * Ph * Pw + (size_t)y * Pw + x0;
#pragma unroll
        for (int r2 = 0; r2 < 8; ++r2) {
            int px = wave * 16 + hi * 8 + r2;
            float u = acc[r2] + bv;
            sb[px] = 1.0f / (1.0f + __expf(-u));
        }
    }
}

// ---------------------------------------------------------------------------
// Final fuse: out = x * (s3[b,h,w] + s1[b,c,w] + s2[b,h,c]) / 3   (v4f).
// x load is last-use and out is never re-read -> non-temporal hints so the
// 134+134 MB stream does not evict L2-resident data.
// ---------------------------------------------------------------------------
__global__ void combine_kernel(const float* __restrict__ x,
                               const float* __restrict__ s3,
                               const float* __restrict__ s1,
                               const float* __restrict__ s2,
                               float* __restrict__ out) {
    size_t i4 = (size_t)blockIdx.x * blockDim.x + threadIdx.x;
    const size_t N = (size_t)BB * CC * HW;
    if (i4 >= N / 4) return;
    size_t i = i4 * 4;

    int b  = (int)(i >> 22);          // C*H*W = 2^22
    int c  = (int)(i >> 16) & 63;
    int h  = (int)(i >> 8) & 255;
    int w  = (int)i & 255;

    // prefetch the streaming operand ahead (global_prefetch_b8 path)
    if (i + 16384 < N) __builtin_prefetch(&x[i + 16384], 0, 1);

    const v4f xv = __builtin_nontemporal_load((const v4f*)(x + i));
    const v4f v3 = *(const v4f*)(s3 + (size_t)b * HW + (size_t)h * WW + w);
    const v4f v1 = *(const v4f*)(s1 + ((size_t)b * CC + c) * WW + w);
    const float v2 = s2[((size_t)b * HH + h) * CC + c];

    const float third = 1.0f / 3.0f;
    v4f o;
    o.x = xv.x * (v3.x + v1.x + v2) * third;
    o.y = xv.y * (v3.y + v1.y + v2) * third;
    o.z = xv.z * (v3.z + v1.z + v2) * third;
    o.w = xv.w * (v3.w + v1.w + v2) * third;
    __builtin_nontemporal_store(o, (v4f*)(out + i));
}

// ---------------------------------------------------------------------------
extern "C" void kernel_launch(void* const* d_in, const int* in_sizes, int n_in,
                              void* d_out, int out_size, void* d_ws, size_t ws_size,
                              hipStream_t stream) {
    const float* x    = (const float*)d_in[0];
    const float* w_cw = (const float*)d_in[1];
    const float* b_cw = (const float*)d_in[2];
    const float* w_hc = (const float*)d_in[3];
    const float* b_hc = (const float*)d_in[4];
    const float* w_hw = (const float*)d_in[5];
    const float* b_hw = (const float*)d_in[6];
    float* out = (float*)d_out;

    // workspace layout (floats): m3 | m1 | m2 | s3 | s1 | s2  (~9.4 MB total)
    float* m3 = (float*)d_ws;                  // [B,2,H,W] = 1048576
    float* m1 = m3 + (size_t)BB * 2 * HW;      // [B,2,C,W] =  262144
    float* m2 = m1 + (size_t)BB * 2 * CC * WW; // [B,2,H,C] =  262144
    float* s3 = m2 + (size_t)BB * 2 * HH * CC; // [B,H,W]   =  524288
    float* s1 = s3 + (size_t)BB * HW;          // [B,C,W]   =  131072
    float* s2 = s1 + (size_t)BB * CC * WW;     // [B,H,C]   =  131072

    // pooled statistics (x becomes L2-resident on first pass)
    reduce_c_kernel<<<(BB * HW) / 256, 256, 0, stream>>>(x, m3);
    reduce_h_kernel<<<(BB * CC * WW) / 256, 256, 0, stream>>>(x, m1);
    reduce_w_kernel<<<(BB * CC * HH * 32) / 256, 256, 0, stream>>>(x, m2);

    // 7x7 conv + sigmoid per branch (WMMA implicit GEMM)
    conv_wmma_kernel<HH, WW, 128><<<BB * HH * (WW / 128), 256, 0, stream>>>(m3, w_hw, b_hw, s3);
    conv_wmma_kernel<CC, WW, 128><<<BB * CC * (WW / 128), 256, 0, stream>>>(m1, w_cw, b_cw, s1);
    conv_wmma_kernel<HH, CC, 64><<<BB * HH * (CC / 64), 128, 0, stream>>>(m2, w_hc, b_hc, s2);

    // fused elementwise combine (v4f streaming, NT hints)
    combine_kernel<<<(int)(((size_t)BB * CC * HW / 4) / 256), 256, 0, stream>>>(x, s3, s1, s2, out);
}
